// VQVAE_Adapter_55284819034580
// MI455X (gfx1250) — matrix operand order repaired and checked by reading, not compiled, not run on hardware
//
#include <hip/hip_runtime.h>
#include <hip/hip_bf16.h>

// ---------- types ----------
typedef __attribute__((ext_vector_type(8)))  float          floatx8;
typedef __attribute__((ext_vector_type(16))) unsigned short ushortx16;
typedef __attribute__((ext_vector_type(16))) __bf16         bf16x16;

#define SERIES 512
#define LSEQ   1024
#define DCH    128
#define PROWS  1026                 // pad rows + 1024 data rows
#define SA_H   (PROWS * DCH)        // 131328 halfs per series (128-ch padded buf)
#define SB64_H (PROWS * 64)         // 65664 halfs per series (64-ch padded buf)
#define NPATCH 64
#define NCODE  1024
#define NVQ    (SERIES * NPATCH)    // 32768 VQ rows

// ---------- scalar helpers ----------
__device__ __forceinline__ unsigned short f2bf(float f) {
    unsigned u = __builtin_bit_cast(unsigned, f);
    unsigned r = u + 0x7FFFu + ((u >> 16) & 1u);
    return (unsigned short)(r >> 16);
}
__device__ __forceinline__ float bf2f(unsigned short h) {
    unsigned u = ((unsigned)h) << 16;
    return __builtin_bit_cast(float, u);
}

// ---------- WMMA fragment loaders (bf16, 16x16x32) ----------
// A (16xK=32, row-major source): lanes 0-15 row M=lane, K halfs [0..8)+[16..24);
// lanes 16-31 row M=lane-16, K halfs [8..16)+[24..32).
__device__ __forceinline__ ushortx16 loadA_g(const unsigned short* base, int lda,
                                             int row0, int kOff, int lane) {
    int r   = lane & 15;
    int sel = (lane < 16) ? 0 : 8;
    const unsigned short* p = base + (long long)(row0 + r) * lda + kOff + sel;
    union { ushortx16 v; uint4 q[2]; } u;
    u.q[0] = *(const uint4*)(p);
    u.q[1] = *(const uint4*)(p + 16);
    return u.v;
}
__device__ __forceinline__ ushortx16 loadA_lds(const unsigned short* sm, int lda,
                                               int row0, int kOff, int lane) {
    int r   = lane & 15;
    int sel = (lane < 16) ? 0 : 8;
    const unsigned short* p = sm + (row0 + r) * lda + kOff + sel;
    union { ushortx16 v; uint4 q[2]; } u;
    u.q[0] = *(const uint4*)(p);
    u.q[1] = *(const uint4*)(p + 16);
    return u.v;
}
// B (K=32 x 16), Bt stored [N][K] row-major: lanes 0-15 col N=lane, K [0..8)+[16..24);
// lanes 16-31 col N=lane-16, K [8..16)+[24..32).
__device__ __forceinline__ ushortx16 loadB_g(const unsigned short* bt, int K,
                                             int nBase, int kOff, int lane) {
    int n   = lane & 15;
    int sel = (lane < 16) ? 0 : 8;
    const unsigned short* p = bt + (long long)(nBase + n) * K + kOff + sel;
    union { ushortx16 v; uint4 q[2]; } u;
    u.q[0] = *(const uint4*)(p);
    u.q[1] = *(const uint4*)(p + 16);
    return u.v;
}
__device__ __forceinline__ floatx8 wmma_bf16(ushortx16 a, ushortx16 b, floatx8 c) {
    return __builtin_amdgcn_wmma_f32_16x16x32_bf16(
        false, __builtin_bit_cast(bf16x16, a),
        false, __builtin_bit_cast(bf16x16, b),
        (short)0, c, false, false);
}

// ---------- generic WMMA GEMM (conv-as-GEMM / patchify / up) ----------
// out(m,n) = sum_k A(m,k)*Bt(n,k), A(m,k) at Aser + m*lda + k  (im2col-free)
struct GemmP {
    const unsigned short* A;   long long sA;  int lda; int K;
    const unsigned short* Bt;  int N;
    int mode;                  // 0 relu, 1 relu+down-res+relu, 2 relu+id-res+relu,
                               // 3 gelu, 4 patchify (f32+bf16 out), 5 up-scatter
    unsigned short* outB;      long long sO;  int outCh; int outPad;
    float* outF;               unsigned short* outH2;
    const float* bias;
    const float* wd; const float* bd; const float* xn;
    int ldsHalfs;              // 0 => load A straight from global
};

template <bool LDSA, int MCOUNT>
__device__ __forceinline__ void gemm_body(const GemmP p, unsigned short* smA) {
    const int tid  = threadIdx.x;
    const int lane = tid & 31;
    const int wid  = tid >> 5;
    const int s    = blockIdx.y;
    const int mbase = blockIdx.x * 64;
    const unsigned short* Aser = p.A + (long long)s * p.sA + (long long)mbase * p.lda;

    if (LDSA) {
        const int chunks = p.ldsHalfs >> 3;           // 16B chunks
        const uint4* g = (const uint4*)Aser;
        uint4* sh = (uint4*)smA;
        for (int i = tid; i < chunks; i += 256) sh[i] = g[i];
        __syncthreads();
    }

    const int ntiles = p.N >> 4;
    int ntStart, mstart;
    if (MCOUNT == 4) { ntStart = wid; mstart = 0; }
    else             { ntStart = wid & (ntiles - 1); mstart = (wid / ntiles) * MCOUNT; }
    const int ksteps = p.K >> 5;

    for (int nt = ntStart; nt < ntiles; nt += 8) {
        floatx8 acc[MCOUNT];
#pragma unroll
        for (int i = 0; i < MCOUNT; ++i) acc[i] = (floatx8)0.0f;

        for (int kk = 0; kk < ksteps; ++kk) {
            const int kOff = kk << 5;
            ushortx16 bfr = loadB_g(p.Bt, p.K, nt << 4, kOff, lane);
#pragma unroll
            for (int mi = 0; mi < MCOUNT; ++mi) {
                const int row = (mstart + mi) << 4;
                ushortx16 afr = LDSA ? loadA_lds(smA, p.lda, row, kOff, lane)
                                     : loadA_g(Aser, p.lda, row, kOff, lane);
                acc[mi] = wmma_bf16(afr, bfr, acc[mi]);
            }
        }
        // epilogue: C/D layout — VGPR r: lanes 0-15 M=r, lanes 16-31 M=r+8; N=lane&15
        const int nloc = lane & 15;
        const int mhi  = (lane >> 4) << 3;
#pragma unroll
        for (int mi = 0; mi < MCOUNT; ++mi) {
#pragma unroll
            for (int r = 0; r < 8; ++r) {
                const int m = mbase + ((mstart + mi) << 4) + r + mhi;
                const int n = (nt << 4) + nloc;
                const float v = acc[mi][r];
                if (p.mode <= 3) {
                    long long o = (long long)s * p.sO + (long long)(p.outPad + m) * p.outCh + n;
                    float y = v + p.bias[n];
                    if (p.mode == 3) {
                        y = 0.5f * y * (1.0f + erff(y * 0.70710678118654752f)); // exact GELU
                    } else {
                        y = fmaxf(y, 0.0f);
                        if (p.mode == 1)
                            y = fmaxf(y + p.wd[n] * p.xn[(long long)s * LSEQ + m] + p.bd[n], 0.0f);
                        else if (p.mode == 2)
                            y = fmaxf(y + bf2f(p.outB[o]), 0.0f);
                    }
                    p.outB[o] = f2bf(y);
                } else if (p.mode == 4) {
                    long long o = (long long)s * (NPATCH * DCH) + (long long)m * DCH + n;
                    float y = v + p.bias[n];
                    p.outF[o]  = y;
                    p.outH2[o] = f2bf(y);
                } else { // mode 5: up-scatter, n = j*128 + d, t = m*16 + j
                    int j = n >> 7, d = n & 127;
                    float y = v + p.bias[d];
                    p.outB[(long long)s * p.sO + (long long)(1 + (m << 4) + j) * DCH + d] = f2bf(y);
                }
            }
        }
    }
}

__global__ __launch_bounds__(256) void gemm_kernel(GemmP p) {
    extern __shared__ unsigned short smA[];
    if (p.N >= 128) {
        if (p.ldsHalfs) gemm_body<true, 4>(p, smA);
        else            gemm_body<false, 4>(p, smA);
    } else {
        if (p.ldsHalfs) gemm_body<true, 2>(p, smA);
        else            gemm_body<false, 2>(p, smA);
    }
}

// ---------- RevIN: per-series mean/var normalize ----------
__global__ __launch_bounds__(256) void revin_kernel(const float* x, const float* rw,
                                                    const float* rb, float* xn) {
    const int s = blockIdx.x, b = s >> 3, c = s & 7, tid = threadIdx.x;
    const float* xs = x + (long long)b * LSEQ * 8 + c;
    float sum = 0.f, sq = 0.f;
    for (int t = tid; t < LSEQ; t += 256) { float v = xs[t * 8]; sum += v; sq += v * v; }
    __shared__ float r1[256], r2[256];
    r1[tid] = sum; r2[tid] = sq; __syncthreads();
    for (int o = 128; o > 0; o >>= 1) {
        if (tid < o) { r1[tid] += r1[tid + o]; r2[tid] += r2[tid + o]; }
        __syncthreads();
    }
    float mu  = r1[0] * (1.0f / LSEQ);
    float var = r2[0] * (1.0f / LSEQ) - mu * mu;
    float scale = rw[0] / sqrtf(var + 1e-5f);
    float shift = rb[0] - mu * scale;
    for (int t = tid; t < LSEQ; t += 256)
        xn[(long long)s * LSEQ + t] = xs[t * 8] * scale + shift;
}

// ---------- TCN block0 conv1 (1->128, k=3 causal) + relu -> bf16 padded ----------
__global__ __launch_bounds__(128) void tcn0_conv1(const float* xn, const float* w1,
                                                  const float* b1, unsigned short* A) {
    const int t = blockIdx.x, s = blockIdx.y, d = threadIdx.x;
    const float* xs = xn + (long long)s * LSEQ;
    float x2 = xs[t];
    float x1 = (t >= 1) ? xs[t - 1] : 0.f;
    float x0 = (t >= 2) ? xs[t - 2] : 0.f;
    float h = w1[d * 3] * x0 + w1[d * 3 + 1] * x1 + w1[d * 3 + 2] * x2 + b1[d];
    A[(long long)s * SA_H + (long long)(2 + t) * DCH + d] = f2bf(fmaxf(h, 0.f));
}

// ---------- weight packers: f32 -> bf16 Bt[N][K] layouts ----------
__global__ void pack_conv_w(const float* w, unsigned short* bt, int cout) {
    int i = blockIdx.x * 256 + threadIdx.x;
    if (i >= cout * 384) return;
    int co = i / 384, k = i % 384, dt = k >> 7, ci = k & 127;
    bt[i] = f2bf(w[(co * 128 + ci) * 3 + dt]);
}
__global__ void pack_proj(const float* w, unsigned short* bt) {
    int i = blockIdx.x * 256 + threadIdx.x;       // 128*2048
    int co = i >> 11, k = i & 2047, j = k >> 7, ci = k & 127;
    bt[i] = f2bf(w[(co * 128 + ci) * 16 + j]);
}
__global__ void pack_up(const float* w, unsigned short* bt) {
    int i = blockIdx.x * 256 + threadIdx.x;       // 2048*128 : Bt[j*128+d][e]
    int n = i >> 7, e = i & 127, j = n >> 7, d = n & 127;
    bt[i] = f2bf(w[(e * 128 + d) * 16 + j]);
}
__global__ void pack_embT(const float* embed, unsigned short* bt) {
    int i = blockIdx.x * 256 + threadIdx.x;       // 1024*128 : Bt[n][d]
    int n = i >> 7, d = i & 127;
    bt[i] = f2bf(embed[d * NCODE + n]);
}
__global__ void enorm_kernel(const float* embed, float* en) {
    int n = blockIdx.x * 256 + threadIdx.x;
    if (n >= NCODE) return;
    float s = 0.f;
    for (int d = 0; d < DCH; ++d) { float e = embed[d * NCODE + n]; s += e * e; }
    en[n] = s;
}

// ---------- pad-row zeroing for buffer reuse ----------
__global__ void zero_rowA(unsigned short* A) {       // grid 512, 128 thr
    A[(long long)blockIdx.x * SA_H + (long long)1025 * DCH + threadIdx.x] = 0;
}
__global__ void zero_rowsB64(unsigned short* B) {    // grid 512, 64 thr
    long long b = (long long)blockIdx.x * SB64_H;
    B[b + threadIdx.x] = 0;
    B[b + (long long)1025 * 64 + threadIdx.x] = 0;
}

// ---------- VQ: argmin over 1024 codes via WMMA dot + LDS scan ----------
__global__ __launch_bounds__(256) void vq_argmin(const unsigned short* pfb,
                                                 const unsigned short* embT,
                                                 const float* enorm, int* ind) {
    __shared__ float sc[16 * 128];
    __shared__ float bV[128];
    __shared__ int   bI[128];
    const int tid = threadIdx.x, lane = tid & 31, wid = tid >> 5;
    const unsigned short* Abase = pfb + (long long)blockIdx.x * 16 * DCH;
    const int row = tid >> 3, seg = tid & 7;      // valid for tid<128
    float bestv = 3.4e38f; int besti = 0;

    for (int p = 0; p < 8; ++p) {                 // 128 codes per pass
        floatx8 acc = (floatx8)0.0f;
        const int nb = p * 128 + wid * 16;
#pragma unroll
        for (int kk = 0; kk < 4; ++kk) {
            ushortx16 a = loadA_g(Abase, DCH, 0, kk * 32, lane);
            ushortx16 b = loadB_g(embT, DCH, nb, kk * 32, lane);
            acc = wmma_bf16(a, b, acc);
        }
#pragma unroll
        for (int r = 0; r < 8; ++r) {
            int m   = r + ((lane >> 4) << 3);
            int col = wid * 16 + (lane & 15);
            sc[m * 128 + col] = enorm[p * 128 + col] - 2.0f * acc[r];
        }
        __syncthreads();
        if (tid < 128) {
            for (int j = 0; j < 16; ++j) {
                int col = seg * 16 + j;
                float v = sc[row * 128 + col];
                if (v < bestv) { bestv = v; besti = p * 128 + col; }
            }
        }
        __syncthreads();
    }
    if (tid < 128) { bV[tid] = bestv; bI[tid] = besti; }
    __syncthreads();
    if (tid < 16) {
        float bv = bV[tid * 8]; int bi = bI[tid * 8];
        for (int g = 1; g < 8; ++g) {
            float v = bV[tid * 8 + g]; int ix = bI[tid * 8 + g];
            if (v < bv || (v == bv && ix < bi)) { bv = v; bi = ix; }
        }
        ind[blockIdx.x * 16 + tid] = bi;
    }
}

// ---------- gather codes -> z_q out (f32 + bf16) + vq-loss partials ----------
__global__ __launch_bounds__(128) void vq_gather(const int* ind, const float* embed,
                                                 const float* pf, float* zq,
                                                 unsigned short* qbf, float* vq_part) {
    const int n = blockIdx.x, d = threadIdx.x;
    const int code = ind[n];
    float e = embed[(long long)d * NCODE + code];
    long long o = (long long)n * DCH + d;
    zq[o]  = e;                     // z_q = pf + sg(q - pf) == q numerically
    qbf[o] = f2bf(e);
    float df = e - pf[o];
    __shared__ float red[128];
    red[d] = df * df; __syncthreads();
    for (int s = 64; s > 0; s >>= 1) {
        if (d < s) red[d] += red[d + s];
        __syncthreads();
    }
    if (d == 0) vq_part[n] = red[0];
}

// ---------- decoder conv2 (64->1, k=3) + recon-loss partials ----------
__global__ __launch_bounds__(256) void dec2_recon(const unsigned short* d1,
                                                  const float* w2, const float* b2,
                                                  const float* xn, float* rec_part) {
    const int s = blockIdx.x, tid = threadIdx.x;
    __shared__ float w[192];
    if (tid < 192) w[tid] = w2[tid];
    __syncthreads();
    const unsigned short* ds = d1 + (long long)s * SB64_H;
    float acc = 0.f;
    for (int t = tid; t < LSEQ; t += 256) {
        float o = b2[0];
#pragma unroll
        for (int dt = 0; dt < 3; ++dt) {
            const unsigned short* rowp = ds + (long long)(t + dt) * 64;
#pragma unroll 8
            for (int ci = 0; ci < 64; ++ci)
                o += bf2f(rowp[ci]) * w[ci * 3 + dt];
        }
        float df = o - xn[(long long)s * LSEQ + t];
        acc += df * df;
    }
    __shared__ float red[256];
    red[tid] = acc; __syncthreads();
    for (int o2 = 128; o2 > 0; o2 >>= 1) {
        if (tid < o2) red[tid] += red[tid + o2];
        __syncthreads();
    }
    if (tid == 0) rec_part[s] = red[0];
}

// ---------- final scalar loss ----------
__global__ __launch_bounds__(256) void finalize_loss(const float* vqp, const float* recp,
                                                     float* out) {
    const int tid = threadIdx.x;
    float a = 0.f, b = 0.f;
    for (int i = tid; i < NVQ; i += 256) a += vqp[i];
    for (int i = tid; i < SERIES; i += 256) b += recp[i];
    __shared__ float r1[256], r2[256];
    r1[tid] = a; r2[tid] = b; __syncthreads();
    for (int o = 128; o > 0; o >>= 1) {
        if (tid < o) { r1[tid] += r1[tid + o]; r2[tid] += r2[tid + o]; }
        __syncthreads();
    }
    if (tid == 0)
        out[(long long)NVQ * DCH] =
            r2[0] / (float)(SERIES * LSEQ) + 0.25f * r1[0] / (float)(NVQ * DCH);
}

// ================= host launcher =================
extern "C" void kernel_launch(void* const* d_in, const int* in_sizes, int n_in,
                              void* d_out, int out_size, void* d_ws, size_t ws_size,
                              hipStream_t stream) {
    (void)in_sizes; (void)n_in; (void)out_size; (void)ws_size;
    const float* x       = (const float*)d_in[0];
    const float* revin_w = (const float*)d_in[1];
    const float* revin_b = (const float*)d_in[2];
    const float* tb0_w1  = (const float*)d_in[3];
    const float* tb0_b1  = (const float*)d_in[4];
    const float* tb0_w2  = (const float*)d_in[5];
    const float* tb0_b2  = (const float*)d_in[6];
    const float* tb0_wd  = (const float*)d_in[7];
    const float* tb0_bd  = (const float*)d_in[8];
    const float* tb1_w1  = (const float*)d_in[9];
    const float* tb1_b1  = (const float*)d_in[10];
    const float* tb1_w2  = (const float*)d_in[11];
    const float* tb1_b2  = (const float*)d_in[12];
    const float* proj_w  = (const float*)d_in[13];
    const float* proj_b  = (const float*)d_in[14];
    const float* embed   = (const float*)d_in[15];
    const float* up_w    = (const float*)d_in[16];
    const float* up_b    = (const float*)d_in[17];
    const float* dec_w1  = (const float*)d_in[18];
    const float* dec_b1  = (const float*)d_in[19];
    const float* dec_w2  = (const float*)d_in[20];
    const float* dec_b2  = (const float*)d_in[21];
    float* out = (float*)d_out;

    // workspace layout (all 256B-aligned)
    char* ws = (char*)d_ws;
    size_t OFF_XN  = 0;                                   // 512*1024 f32
    size_t OFF_A   = OFF_XN  + 2097152;                   // 512*131328 bf16
    size_t OFF_B   = OFF_A   + 134479872;
    size_t OFF_PF  = OFF_B   + 134479872;                 // pf f32
    size_t OFF_PFB = OFF_PF  + 16777216;                  // pf bf16
    size_t OFF_QB  = OFF_PFB + 8388608;                   // q bf16
    size_t OFF_ET  = OFF_QB  + 8388608;                   // embT bf16
    size_t OFF_EN  = OFF_ET  + 262144;                    // enorm f32
    size_t OFF_IND = OFF_EN  + 4096;                      // ind i32
    size_t OFF_W02 = OFF_IND + 131072;
    size_t OFF_W11 = OFF_W02 + 98304;
    size_t OFF_W12 = OFF_W11 + 98304;
    size_t OFF_WPJ = OFF_W12 + 98304;
    size_t OFF_WUP = OFF_WPJ + 524288;
    size_t OFF_WD1 = OFF_WUP + 524288;
    size_t OFF_VQP = OFF_WD1 + 49152;
    size_t OFF_RCP = OFF_VQP + 131072;

    float*          xn   = (float*)(ws + OFF_XN);
    unsigned short* Abuf = (unsigned short*)(ws + OFF_A);
    unsigned short* Bbuf = (unsigned short*)(ws + OFF_B);
    float*          pf   = (float*)(ws + OFF_PF);
    unsigned short* pfb  = (unsigned short*)(ws + OFF_PFB);
    unsigned short* qbf  = (unsigned short*)(ws + OFF_QB);
    unsigned short* embT = (unsigned short*)(ws + OFF_ET);
    float*          en   = (float*)(ws + OFF_EN);
    int*            ind  = (int*)(ws + OFF_IND);
    unsigned short* w02  = (unsigned short*)(ws + OFF_W02);
    unsigned short* w11  = (unsigned short*)(ws + OFF_W11);
    unsigned short* w12  = (unsigned short*)(ws + OFF_W12);
    unsigned short* wpj  = (unsigned short*)(ws + OFF_WPJ);
    unsigned short* wup  = (unsigned short*)(ws + OFF_WUP);
    unsigned short* wd1  = (unsigned short*)(ws + OFF_WD1);
    float*          vqp  = (float*)(ws + OFF_VQP);
    float*          rcp  = (float*)(ws + OFF_RCP);

    // zero pad rows of the two big activation buffers (bf16 zero == 0x0000)
    hipMemsetAsync(Abuf, 0, 134479872, stream);
    hipMemsetAsync(Bbuf, 0, 134479872, stream);

    // pack weights to bf16 Bt layouts + code norms
    pack_conv_w<<<192,  256, 0, stream>>>(tb0_w2, w02, 128);
    pack_conv_w<<<192,  256, 0, stream>>>(tb1_w1, w11, 128);
    pack_conv_w<<<192,  256, 0, stream>>>(tb1_w2, w12, 128);
    pack_conv_w<<<96,   256, 0, stream>>>(dec_w1, wd1, 64);
    pack_proj  <<<1024, 256, 0, stream>>>(proj_w, wpj);
    pack_up    <<<1024, 256, 0, stream>>>(up_w,   wup);
    pack_embT  <<<512,  256, 0, stream>>>(embed,  embT);
    enorm_kernel<<<4,   256, 0, stream>>>(embed,  en);

    // RevIN + block0 conv1
    revin_kernel<<<SERIES, 256, 0, stream>>>(x, revin_w, revin_b, xn);
    tcn0_conv1<<<dim3(LSEQ, SERIES), 128, 0, stream>>>(xn, tb0_w1, tb0_b1, Abuf);

    GemmP p{};
    auto launch = [&](int mtiles, size_t shmem) {
        gemm_kernel<<<dim3(mtiles, SERIES), 256, shmem, stream>>>(p);
    };
    const int LDS_CONV = 63 * 128 + 384;   // 8448 halfs
    const int LDS_UP   = 63 * 128 + 128;   // 8192 halfs

    // block0 conv2: relu(relu(conv)+downsample(xn))
    p = GemmP{Abuf, SA_H, 128, 384, w02, 128, 1, Bbuf, SA_H, DCH, 2,
              nullptr, nullptr, tb0_b2, tb0_wd, tb0_bd, xn, LDS_CONV};
    launch(16, (size_t)LDS_CONV * 2);
    // block1 conv1: relu(conv)
    p = GemmP{Bbuf, SA_H, 128, 384, w11, 128, 0, Abuf, SA_H, DCH, 2,
              nullptr, nullptr, tb1_b1, nullptr, nullptr, nullptr, LDS_CONV};
    launch(16, (size_t)LDS_CONV * 2);
    // block1 conv2: relu(relu(conv)+res) in-place into Bbuf (res read per element)
    p = GemmP{Abuf, SA_H, 128, 384, w12, 128, 2, Bbuf, SA_H, DCH, 2,
              nullptr, nullptr, tb1_b2, nullptr, nullptr, nullptr, LDS_CONV};
    launch(16, (size_t)LDS_CONV * 2);
    // patchify: lda=K=2048 GEMM, f32+bf16 outputs (A staged in L2, too big for LDS tile)
    p = GemmP{Bbuf + 256, SA_H, 2048, 2048, wpj, 128, 4, nullptr, 0, 0, 0,
              pf, pfb, proj_b, nullptr, nullptr, nullptr, 0};
    launch(1, 0);

    // VQ
    vq_argmin<<<NVQ / 16, 256, 0, stream>>>(pfb, embT, en, ind);
    vq_gather<<<NVQ, 128, 0, stream>>>(ind, embed, pf, out, qbf, vqp);

    // decoder: up-projection (ConvTranspose k=s=16) into Abuf (pad-1 layout)
    zero_rowA<<<SERIES, 128, 0, stream>>>(Abuf);
    p = GemmP{qbf, (long long)NPATCH * DCH, 128, 128, wup, 2048, 5, Abuf, SA_H, DCH, 1,
              nullptr, nullptr, up_b, nullptr, nullptr, nullptr, LDS_UP};
    launch(1, (size_t)LDS_UP * 2);
    // dec conv1 (128->64, pad 1) + exact GELU, into Bbuf reinterpreted [s][1026][64]
    zero_rowsB64<<<SERIES, 64, 0, stream>>>(Bbuf);
    p = GemmP{Abuf, SA_H, 128, 384, wd1, 64, 3, Bbuf, SB64_H, 64, 1,
              nullptr, nullptr, dec_b1, nullptr, nullptr, nullptr, LDS_CONV};
    launch(16, (size_t)LDS_CONV * 2);
    // dec conv2 (64->1) + recon partials, then scalar loss
    dec2_recon<<<SERIES, 256, 0, stream>>>(Bbuf, dec_w2, dec_b2, xn, rcp);
    finalize_loss<<<1, 256, 0, stream>>>(vqp, rcp, out);
}
